// SelectiveTemporalStateSpace_13365938225637
// MI455X (gfx1250) — matrix operand order repaired and checked by reading, not compiled, not run on hardware
//
#include <hip/hip_runtime.h>
#include <hip/hip_bf16.h>

// Problem constants (reference: B=16, L=1024, D=1024)
#define BB 16
#define LL 1024
#define DD 1024
#define NPROJ (3 * DD)          // 3072
#define MROWS (BB * LL)         // 16384
#define EPS 1e-5f

// Chunked scan parameters: LL = NCHUNK * CHLEN
#define NCHUNK 16
#define CHLEN 64
#define NSEQ (BB * DD)          // 16384 independent recurrences

typedef __bf16 v16bf __attribute__((ext_vector_type(16)));
typedef float  v8f   __attribute__((ext_vector_type(8)));

union alignas(32) Frag {
    v16bf v;
    uint4 q[2];
};

__device__ __forceinline__ unsigned short f2bf(float f) {
    unsigned int u = __float_as_uint(f);
    unsigned int r = (u + 0x7fffu + ((u >> 16) & 1u)) >> 16;
    return (unsigned short)r;
}
__device__ __forceinline__ float bf2f(unsigned short h) {
    return __uint_as_float(((unsigned int)h) << 16);
}

// ---------------------------------------------------------------------------
// K0: LayerNorm over D, write xn as bf16 (row-major [MROWS][DD])
// ---------------------------------------------------------------------------
__global__ __launch_bounds__(256)
void ln_kernel(const float* __restrict__ x, const float* __restrict__ gamma,
               const float* __restrict__ beta, unsigned short* __restrict__ xn) {
    const int row = blockIdx.x;
    const float* xr = x + (size_t)row * DD;
    float v[4];
    float s = 0.f, sq = 0.f;
#pragma unroll
    for (int i = 0; i < 4; ++i) {
        v[i] = xr[threadIdx.x + i * 256];
        s += v[i];
        sq += v[i] * v[i];
    }
#pragma unroll
    for (int off = 16; off > 0; off >>= 1) {
        s  += __shfl_down(s, off, 32);
        sq += __shfl_down(sq, off, 32);
    }
    __shared__ float ss[8], ssq[8];
    const int w = threadIdx.x >> 5, ln = threadIdx.x & 31;
    if (ln == 0) { ss[w] = s; ssq[w] = sq; }
    __syncthreads();
    if (threadIdx.x == 0) {
        float a = 0.f, b = 0.f;
#pragma unroll
        for (int i = 0; i < 8; ++i) { a += ss[i]; b += ssq[i]; }
        ss[0] = a; ssq[0] = b;
    }
    __syncthreads();
    const float mean = ss[0] * (1.f / DD);
    const float var  = ssq[0] * (1.f / DD) - mean * mean;
    const float rs   = rsqrtf(var + EPS);
#pragma unroll
    for (int i = 0; i < 4; ++i) {
        const int d = threadIdx.x + i * 256;
        const float y = (v[i] - mean) * rs * gamma[d] + beta[d];
        xn[(size_t)row * DD + d] = f2bf(y);
    }
}

// ---------------------------------------------------------------------------
// K1: pack W_proj [DD][NPROJ] fp32 -> bf16 in WMMA B-fragment order.
//   k = tile_k*32 + (e<8 ? (lane>>4)*8 + e : 16 + (lane>>4)*8 + (e-8)),  n = tile_n*16 + (lane&15)
//   flat index = ((tile_n*32 + tile_k)*32 + lane)*16 + e
// ---------------------------------------------------------------------------
__global__ __launch_bounds__(256)
void packw_kernel(const float* __restrict__ W, unsigned short* __restrict__ wpk) {
    const size_t g = (size_t)blockIdx.x * 256 + threadIdx.x;   // < DD*NPROJ
    const int e      = (int)(g & 15);
    const int lane   = (int)((g >> 4) & 31);
    const size_t t   = g >> 9;
    const int tile_k = (int)(t & 31);        // DD/32 = 32 k-tiles
    const int tile_n = (int)(t >> 5);        // NPROJ/16 = 192 n-tiles
    const int k = tile_k * 32 + ((e < 8) ? ((lane >> 4) * 8 + e) : (16 + (lane >> 4) * 8 + (e - 8)));
    const int n = tile_n * 16 + (lane & 15);
    wpk[g] = f2bf(W[(size_t)k * NPROJ + n]);
}

// ---------------------------------------------------------------------------
// K2: depthwise conv1d(k=3, pad=1) along L on xn, sigmoid -> gate fp32
// ---------------------------------------------------------------------------
__global__ __launch_bounds__(256)
void gate_kernel(const unsigned short* __restrict__ xn, const float* __restrict__ cw,
                 const float* __restrict__ cb, float* __restrict__ gate) {
    const size_t g = (size_t)blockIdx.x * 256 + threadIdx.x;   // < MROWS*DD
    const int d = (int)(g & (DD - 1));
    const size_t row = g >> 10;            // row = b*LL + l
    const int l = (int)(row & (LL - 1));
    const float xm = bf2f(xn[g]);
    const float xl = (l > 0)        ? bf2f(xn[g - DD]) : 0.f;
    const float xr = (l < LL - 1)   ? bf2f(xn[g + DD]) : 0.f;
    const float v = cw[d * 3 + 0] * xl + cw[d * 3 + 1] * xm + cw[d * 3 + 2] * xr + cb[d];
    gate[g] = 1.f / (1.f + expf(-v));
}

// ---------------------------------------------------------------------------
// K3: bf16 WMMA GEMM  proj = xn @ W + b  with fused SSM epilogue.
// Block: 256 threads = 8 waves, block tile 128(M) x 128(N),
// wave tile 32(M) x 64(N) = 2x4 WMMA 16x16 tiles, K-loop of DD/32 = 32.
// ---------------------------------------------------------------------------
__global__ __launch_bounds__(256)
void gemm_ssm_kernel(const unsigned short* __restrict__ xn_u,
                     const unsigned short* __restrict__ wpk_u,
                     const float* __restrict__ bproj,
                     const float* __restrict__ gate,
                     const float* __restrict__ xin,
                     const float* __restrict__ ts,
                     float* __restrict__ Abuf,
                     float* __restrict__ Bxbuf,
                     float* __restrict__ Cbuf) {
    const __bf16* xn  = (const __bf16*)xn_u;
    const __bf16* wpk = (const __bf16*)wpk_u;

    const int lane = threadIdx.x & 31;
    const int wave = threadIdx.x >> 5;
    const int m0 = blockIdx.y * 128 + (wave >> 1) * 32;   // 4 waves along M
    const int n0 = blockIdx.x * 128 + (wave & 1) * 64;    // 2 waves along N
    const int half = lane >> 4;
    const int mrow = lane & 15;

    v8f acc[2][4];
#pragma unroll
    for (int i = 0; i < 2; ++i)
#pragma unroll
        for (int j = 0; j < 4; ++j)
            acc[i][j] = (v8f){0.f, 0.f, 0.f, 0.f, 0.f, 0.f, 0.f, 0.f};

    const __bf16* arow0 = xn + (size_t)(m0 + mrow) * DD + 8 * half;
    const __bf16* arow1 = arow0 + (size_t)16 * DD;
    const size_t bbase = (size_t)(n0 >> 4) * 16384 + (size_t)lane * 16;

#pragma unroll 4
    for (int kt = 0; kt < DD / 32; ++kt) {
        const int k0 = kt * 32;
        Frag a[2], b[4];
        a[0].q[0] = *(const uint4*)(arow0 + k0);
        a[0].q[1] = *(const uint4*)(arow0 + k0 + 16);
        a[1].q[0] = *(const uint4*)(arow1 + k0);
        a[1].q[1] = *(const uint4*)(arow1 + k0 + 16);
#pragma unroll
        for (int j = 0; j < 4; ++j) {
            const __bf16* bp = wpk + bbase + (size_t)j * 16384 + (size_t)kt * 512;
            b[j].q[0] = *(const uint4*)(bp);
            b[j].q[1] = *(const uint4*)(bp + 8);
        }
#pragma unroll
        for (int i = 0; i < 2; ++i)
#pragma unroll
            for (int j = 0; j < 4; ++j)
                acc[i][j] = __builtin_amdgcn_wmma_f32_16x16x32_bf16(
                    false, a[i].v, false, b[j].v, (short)0, acc[i][j], false, false);
    }

    // Fused epilogue: bias + segment-specific SSM transform
    const int nn = lane & 15;
#pragma unroll
    for (int i = 0; i < 2; ++i) {
#pragma unroll
        for (int j = 0; j < 4; ++j) {
            const int col = n0 + j * 16 + nn;
            const int seg = col >> 10;          // 0: delta, 1: B, 2: C
            const int d   = col & (DD - 1);
            const float bias = bproj[col];
#pragma unroll
            for (int r = 0; r < 8; ++r) {
                const int row = m0 + i * 16 + (half << 3) + r;
                const float v = acc[i][j][r] + bias;
                const size_t gi = ((size_t)row << 10) + d;
                const float gt = gate[gi];
                if (seg == 0) {
                    const float sp = (v > 20.f) ? v : log1pf(expf(v));
                    const int l = row & (LL - 1);   // time_scale indexed by l (torch broadcast)
                    Abuf[gi] = expf(-sp * gt * expf(ts[l]));
                } else if (seg == 1) {
                    Bxbuf[gi] = v * gt * xin[gi];
                } else {
                    Cbuf[gi] = v * gt;
                }
            }
        }
    }
}

// ---------------------------------------------------------------------------
// Chunked linear recurrence: h_t = A_t * h_{t-1} + Bx_t  over L, per (b,d).
// Decompose into NCHUNK affine maps to expose NSEQ*NCHUNK-way parallelism
// (8192 waves instead of 512 -> HBM-saturating).
// ---------------------------------------------------------------------------
// S1: per-chunk summary (P = prod A, S = h at chunk end with h_in = 0)
__global__ __launch_bounds__(256)
void scan1_kernel(const float* __restrict__ A, const float* __restrict__ Bx,
                  float* __restrict__ P, float* __restrict__ S) {
    const int g = blockIdx.x * 256 + threadIdx.x;   // < NSEQ*NCHUNK
    const int c = g >> 14;                          // chunk index
    const int r = g & (NSEQ - 1);                   // r = b*DD + d
    const int b = r >> 10;
    const int d = r & (DD - 1);
    const size_t base = (size_t)b * LL * DD + (size_t)(c * CHLEN) * DD + d;
    float p = 1.f, s = 0.f;
    for (int l = 0; l < CHLEN; ++l) {
        const size_t idx = base + (size_t)l * DD;
        const float a = A[idx];
        s = a * s + Bx[idx];
        p *= a;
    }
    P[g] = p;
    S[g] = s;
}

// S2: sequentially compose the NCHUNK summaries per sequence (tiny pass)
__global__ __launch_bounds__(256)
void scan2_kernel(const float* __restrict__ P, const float* __restrict__ S,
                  float* __restrict__ Hin) {
    const int r = blockIdx.x * 256 + threadIdx.x;   // < NSEQ
    float h = 0.f;
#pragma unroll
    for (int c = 0; c < NCHUNK; ++c) {
        const int idx = c * NSEQ + r;
        Hin[idx] = h;
        h = P[idx] * h + S[idx];
    }
}

// S3: replay each chunk from its correct incoming state, emit out = x + C*h
__global__ __launch_bounds__(256)
void scan3_kernel(const float* __restrict__ A, const float* __restrict__ Bx,
                  const float* __restrict__ C, const float* __restrict__ x,
                  const float* __restrict__ Hin, float* __restrict__ out) {
    const int g = blockIdx.x * 256 + threadIdx.x;   // < NSEQ*NCHUNK
    const int c = g >> 14;
    const int r = g & (NSEQ - 1);
    const int b = r >> 10;
    const int d = r & (DD - 1);
    const size_t base = (size_t)b * LL * DD + (size_t)(c * CHLEN) * DD + d;
    float h = Hin[g];
    for (int l = 0; l < CHLEN; ++l) {
        const size_t idx = base + (size_t)l * DD;
        h = A[idx] * h + Bx[idx];
        out[idx] = x[idx] + C[idx] * h;
    }
}

// ---------------------------------------------------------------------------
extern "C" void kernel_launch(void* const* d_in, const int* in_sizes, int n_in,
                              void* d_out, int out_size, void* d_ws, size_t ws_size,
                              hipStream_t stream) {
    const float* x        = (const float*)d_in[0];
    const float* ln_gamma = (const float*)d_in[1];
    const float* ln_beta  = (const float*)d_in[2];
    const float* W_proj   = (const float*)d_in[3];
    const float* b_proj   = (const float*)d_in[4];
    const float* conv_w   = (const float*)d_in[5];
    const float* conv_b   = (const float*)d_in[6];
    const float* tscale   = (const float*)d_in[7];
    float* out = (float*)d_out;

    char* ws = (char*)d_ws;
    // workspace layout (offsets in bytes)
    unsigned short* xn   = (unsigned short*)(ws);                              // 32 MB
    unsigned short* wpk  = (unsigned short*)(ws + (size_t)33554432);           //  6 MB
    float*          gate = (float*)(ws + (size_t)39845888);                    // 64 MB
    float*          Abuf = (float*)(ws + (size_t)106954752);                   // 64 MB
    float*          Bxb  = (float*)(ws + (size_t)174063616);                   // 64 MB
    float*          Cbuf = (float*)(ws + (size_t)241172480);                   // 64 MB
    float*          Pbuf = (float*)(ws + (size_t)308281344);                   //  1 MB
    float*          Sbuf = (float*)(ws + (size_t)309329920);                   //  1 MB
    float*          Hin  = (float*)(ws + (size_t)310378496);                   //  1 MB

    // K0: LayerNorm -> xn (bf16)
    ln_kernel<<<MROWS, 256, 0, stream>>>(x, ln_gamma, ln_beta, xn);

    // K1: pack weights into WMMA-B fragment order (bf16)
    packw_kernel<<<(DD * NPROJ) / 256, 256, 0, stream>>>(W_proj, wpk);

    // K2: conv gate
    gate_kernel<<<(MROWS * DD) / 256, 256, 0, stream>>>(xn, conv_w, conv_b, gate);

    // K3: WMMA GEMM + fused SSM epilogue
    dim3 grid(NPROJ / 128, MROWS / 128);   // (24, 128)
    gemm_ssm_kernel<<<grid, 256, 0, stream>>>(xn, wpk, b_proj, gate, x, tscale,
                                              Abuf, Bxb, Cbuf);

    // K4a/b/c: chunked parallel scan + residual add
    scan1_kernel<<<(NSEQ * NCHUNK) / 256, 256, 0, stream>>>(Abuf, Bxb, Pbuf, Sbuf);
    scan2_kernel<<<NSEQ / 256, 256, 0, stream>>>(Pbuf, Sbuf, Hin);
    scan3_kernel<<<(NSEQ * NCHUNK) / 256, 256, 0, stream>>>(Abuf, Bxb, Cbuf, x, Hin, out);
}
